// PhysicsGuidedBlockMultiplicative_29497835389572
// MI455X (gfx1250) — matrix operand order repaired and checked by reading, not compiled, not run on hardware
//
#include <hip/hip_runtime.h>
#include <hip/hip_bf16.h>
#include <math.h>

#define DIMC   768
#define NTOK   2048
#define HEADS  12
#define HEAD_D 64
#define HIDDEN 3072
#define C3     2304

typedef __attribute__((ext_vector_type(16))) __bf16 v16bf;
typedef __attribute__((ext_vector_type(8)))  __bf16 bf16x8;
typedef __attribute__((ext_vector_type(8)))  float  v8f;

union FragBF { v16bf v; bf16x8 h[2]; };

__device__ inline v8f v8f_zero() {
  v8f z;
#pragma unroll
  for (int i = 0; i < 8; ++i) z[i] = 0.f;
  return z;
}

__device__ inline v8f wmma_bf16(v16bf a, v16bf b, v8f c) {
  return __builtin_amdgcn_wmma_f32_16x16x32_bf16(false, a, false, b, (short)0, c, false, false);
}

// log(sigmoid(z)) = -softplus(-z) = -(max(-z,0) + log(1+exp(-|z|)))
// exp argument always <= 0 -> hardware v_exp_f32 needs no range fixup.
__device__ inline float fast_log_sigmoid(float z) {
  float t = __expf(-fabsf(z));
  return -(fmaxf(-z, 0.f) + __logf(1.f + t));
}

// ---------------- fp32 -> bf16 weight conversion ----------------
__global__ void cvt_f32_bf16(const float* __restrict__ in, __bf16* __restrict__ out, int n) {
  int i = blockIdx.x * blockDim.x + threadIdx.x;
  int stride = gridDim.x * blockDim.x;
  for (; i < n; i += stride) out[i] = (__bf16)in[i];
}

// ---------------- LayerNorm (row per block), fp32 in -> bf16 out ----------------
__global__ void layernorm_kernel(const float* __restrict__ x, const float* __restrict__ g,
                                 const float* __restrict__ b, __bf16* __restrict__ out) {
  __shared__ float red[256];
  const int row = blockIdx.x;
  const float* xr = x + (size_t)row * DIMC;
  float s = 0.f;
  for (int i = threadIdx.x; i < DIMC; i += 256) s += xr[i];
  red[threadIdx.x] = s; __syncthreads();
  for (int off = 128; off > 0; off >>= 1) {
    if (threadIdx.x < off) red[threadIdx.x] += red[threadIdx.x + off];
    __syncthreads();
  }
  const float mu = red[0] / DIMC;
  __syncthreads();
  float v = 0.f;
  for (int i = threadIdx.x; i < DIMC; i += 256) { float d = xr[i] - mu; v += d * d; }
  red[threadIdx.x] = v; __syncthreads();
  for (int off = 128; off > 0; off >>= 1) {
    if (threadIdx.x < off) red[threadIdx.x] += red[threadIdx.x + off];
    __syncthreads();
  }
  const float rstd = rsqrtf(red[0] / DIMC + 1e-5f);
  for (int i = threadIdx.x; i < DIMC; i += 256)
    out[(size_t)row * DIMC + i] = (__bf16)((xr[i] - mu) * rstd * g[i] + b[i]);
}

// ---------------- WMMA GEMM: Out[M,N] = A[M,K] @ Bw[N,K]^T + bias (+epilogue) ----
// Block tile 128x128 (8 waves); each wave: 32 rows x 64 cols (8 WMMAs / k-step).
// EPI 0: bf16 out; EPI 1: bf16 out with exact GELU; EPI 2: fp32 out + residual
template <int EPI>
__global__ void gemm_wmma(const __bf16* __restrict__ A, const __bf16* __restrict__ Bw,
                          const float* __restrict__ bias, const float* __restrict__ res,
                          void* __restrict__ Out, int M, int N, int K) {
  const int lane = threadIdx.x & 31;
  const int wave = threadIdx.x >> 5;
  const int wrow = wave & 3, wcol = wave >> 2;
  const int row0 = blockIdx.x * 128 + wrow * 32;
  const int col0 = blockIdx.y * 128 + wcol * 64;
  const int l15 = lane & 15, lh = lane >> 4;
  const int akb = lh * 8;    // A frag: elements 0..7 at +akb, 8..15 at +akb+16
  const int bkb = lh * 8;    // B frag: elements 0..7 at +bkb, 8..15 at +bkb+16
  const __bf16* arp0 = A + (size_t)(row0 + l15) * K + akb;
  const __bf16* arp1 = A + (size_t)(row0 + 16 + l15) * K + akb;

  v8f acc[2][4];
#pragma unroll
  for (int g = 0; g < 2; ++g)
#pragma unroll
    for (int c = 0; c < 4; ++c) acc[g][c] = v8f_zero();

  for (int k0 = 0; k0 < K; k0 += 32) {
    FragBF a0, a1;
    a0.h[0] = *(const bf16x8*)(arp0 + k0);
    a0.h[1] = *(const bf16x8*)(arp0 + k0 + 16);
    a1.h[0] = *(const bf16x8*)(arp1 + k0);
    a1.h[1] = *(const bf16x8*)(arp1 + k0 + 16);
    __builtin_prefetch(arp0 + k0 + 32, 0, 3);  // speculative: dropped if OOB
    __builtin_prefetch(arp1 + k0 + 32, 0, 3);
#pragma unroll
    for (int c = 0; c < 4; ++c) {
      FragBF bf;
      const __bf16* bp = Bw + (size_t)(col0 + c * 16 + l15) * K + k0 + bkb;
      bf.h[0] = *(const bf16x8*)(bp);
      bf.h[1] = *(const bf16x8*)(bp + 16);
      __builtin_prefetch(bp + 32, 0, 3);
      acc[0][c] = wmma_bf16(a0.v, bf.v, acc[0][c]);
      acc[1][c] = wmma_bf16(a1.v, bf.v, acc[1][c]);
    }
  }

  const int rbase = 8 * lh;
#pragma unroll
  for (int g = 0; g < 2; ++g) {
#pragma unroll
    for (int c = 0; c < 4; ++c) {
      const int col = col0 + c * 16 + l15;
      const float bcol = bias[col];
#pragma unroll
      for (int r = 0; r < 8; ++r) {
        const int row = row0 + g * 16 + rbase + r;
        float v = acc[g][c][r] + bcol;
        if (EPI == 0) {
          ((__bf16*)Out)[(size_t)row * N + col] = (__bf16)v;
        } else if (EPI == 1) {
          float ge = 0.5f * v * (1.0f + erff(v * 0.70710678118f));
          ((__bf16*)Out)[(size_t)row * N + col] = (__bf16)ge;
        } else {
          ((float*)Out)[(size_t)row * N + col] = v + res[(size_t)row * N + col];
        }
      }
    }
  }
}

// ---------------- Flash attention with additive log-sigmoid elevation bias -----
// qkv: [NTOK, 3*DIMC] bf16 (q | k | v). out: [NTOK, DIMC] bf16.
// attn = softmax(q k^T/8 + log(clip(sigmoid(-s*(e_j-e_i))))), streamed over keys.
__global__ void flash_attn_kernel(const __bf16* __restrict__ qkv, const float* __restrict__ elev,
                                  const float* __restrict__ strength_p, __bf16* __restrict__ out) {
  __shared__ __align__(64) __bf16 vt[8][64][32];  // per-wave V tile, transposed [d][key]
  __shared__ __align__(64) __bf16 pt[8][16][32];  // per-wave P tile [row][key]
  const int lane = threadIdx.x & 31;
  const int w = threadIdx.x >> 5;
  const int head = blockIdx.y;
  const int qrow0 = (blockIdx.x * 8 + w) * 16;
  const float strength = strength_p[0];
  const float scale = 0.125f;  // 64^-0.5
  const int hoff = head * HEAD_D;
  const int l15 = lane & 15, lh = lane >> 4;
  const float LOGMIN = -13.815511f;     // log(1e-6)
  const float LOGMAX = -1.0000005e-6f;  // log(1-1e-6)

  // Q fragments (16 rows x d: [0,32) and [32,64))
  FragBF aq[2];
  {
    const __bf16* qp = qkv + (size_t)(qrow0 + l15) * C3 + hoff + lh * 16;
    aq[0].h[0] = *(const bf16x8*)(qp);
    aq[0].h[1] = *(const bf16x8*)(qp + 8);
    aq[1].h[0] = *(const bf16x8*)(qp + 32);
    aq[1].h[1] = *(const bf16x8*)(qp + 40);
  }
  float eq[8];
#pragma unroll
  for (int r = 0; r < 8; ++r) eq[r] = elev[qrow0 + 8 * lh + r];

  v8f o[4];
#pragma unroll
  for (int c = 0; c < 4; ++c) o[c] = v8f_zero();
  float m_i[8], l_i[8];
#pragma unroll
  for (int r = 0; r < 8; ++r) { m_i[r] = -1e30f; l_i[r] = 0.f; }

  for (int kb = 0; kb < NTOK; kb += 32) {
    // stage V tile (32 keys x 64 dims) transposed into per-wave LDS
    {
      const bf16x8* vp = (const bf16x8*)(qkv + (size_t)(kb + lane) * C3 + 2 * DIMC + hoff);
#pragma unroll
      for (int ch = 0; ch < 8; ++ch) {
        bf16x8 vv = vp[ch];
#pragma unroll
        for (int e = 0; e < 8; ++e) vt[w][ch * 8 + e][lane] = vv[e];
      }
    }
    const float ek0 = elev[kb + l15];
    const float ek1 = elev[kb + 16 + l15];

    // S = Q K^T for two key groups of 16
    v8f s0 = v8f_zero(), s1 = v8f_zero();
    {
      FragBF kf;
      const __bf16* kpb = qkv + (size_t)(kb + l15) * C3 + DIMC + hoff + lh * 16;
      kf.h[0] = *(const bf16x8*)(kpb);      kf.h[1] = *(const bf16x8*)(kpb + 8);
      s0 = wmma_bf16(aq[0].v, kf.v, s0);
      kf.h[0] = *(const bf16x8*)(kpb + 32); kf.h[1] = *(const bf16x8*)(kpb + 40);
      s0 = wmma_bf16(aq[1].v, kf.v, s0);
      const __bf16* kpb1 = kpb + (size_t)16 * C3;
      kf.h[0] = *(const bf16x8*)(kpb1);      kf.h[1] = *(const bf16x8*)(kpb1 + 8);
      s1 = wmma_bf16(aq[0].v, kf.v, s1);
      kf.h[0] = *(const bf16x8*)(kpb1 + 32); kf.h[1] = *(const bf16x8*)(kpb1 + 40);
      s1 = wmma_bf16(aq[1].v, kf.v, s1);
    }

    // bias + online softmax update (rows split across 16-lane halves)
    float p0[8], p1[8], mx[8];
#pragma unroll
    for (int r = 0; r < 8; ++r) {
      float z0 = -strength * (ek0 - eq[r]);
      float lm0 = fminf(fmaxf(fast_log_sigmoid(z0), LOGMIN), LOGMAX);
      float z1 = -strength * (ek1 - eq[r]);
      float lm1 = fminf(fmaxf(fast_log_sigmoid(z1), LOGMIN), LOGMAX);
      p0[r] = s0[r] * scale + lm0;
      p1[r] = s1[r] * scale + lm1;
      mx[r] = fmaxf(p0[r], p1[r]);
    }
#pragma unroll
    for (int r = 0; r < 8; ++r) {
#pragma unroll
      for (int msk = 1; msk < 16; msk <<= 1) mx[r] = fmaxf(mx[r], __shfl_xor(mx[r], msk, 32));
      float mn = fmaxf(m_i[r], mx[r]);
      float alpha = __expf(m_i[r] - mn);   // arg <= 0
      m_i[r] = mn;
      p0[r] = __expf(p0[r] - mn);          // arg <= 0
      p1[r] = __expf(p1[r] - mn);          // arg <= 0
      float ps = p0[r] + p1[r];
#pragma unroll
      for (int msk = 1; msk < 16; msk <<= 1) ps += __shfl_xor(ps, msk, 32);
      l_i[r] = l_i[r] * alpha + ps;
      o[0][r] *= alpha; o[1][r] *= alpha; o[2][r] *= alpha; o[3][r] *= alpha;
      pt[w][8 * lh + r][l15]      = (__bf16)p0[r];
      pt[w][8 * lh + r][16 + l15] = (__bf16)p1[r];
    }
    asm volatile("s_wait_dscnt 0" ::: "memory");  // same-wave DS ordering barrier

    // O += P (16x32) @ V (32x64), V read transposed from LDS
    FragBF ap;
    ap.h[0] = *(const bf16x8*)&pt[w][l15][lh * 16];
    ap.h[1] = *(const bf16x8*)&pt[w][l15][lh * 16 + 8];
#pragma unroll
    for (int c = 0; c < 4; ++c) {
      FragBF vf;
      const __bf16* vtp = &vt[w][c * 16 + l15][lh * 16];
      vf.h[0] = *(const bf16x8*)(vtp);
      vf.h[1] = *(const bf16x8*)(vtp + 8);
      o[c] = wmma_bf16(ap.v, vf.v, o[c]);
    }
  }

#pragma unroll
  for (int c = 0; c < 4; ++c) {
#pragma unroll
    for (int r = 0; r < 8; ++r) {
      const int row = qrow0 + 8 * lh + r;
      out[(size_t)row * DIMC + hoff + c * 16 + l15] = (__bf16)(o[c][r] / (l_i[r] + 1e-8f));
    }
  }
}

// ---------------------------------------------------------------------------
extern "C" void kernel_launch(void* const* d_in, const int* in_sizes, int n_in,
                              void* d_out, int out_size, void* d_ws, size_t ws_size,
                              hipStream_t stream) {
  (void)in_sizes; (void)n_in; (void)out_size; (void)ws_size;
  const float* x        = (const float*)d_in[0];
  const float* elev     = (const float*)d_in[1];
  const float* n1g      = (const float*)d_in[2];
  const float* n1b      = (const float*)d_in[3];
  const float* qkv_w    = (const float*)d_in[4];
  const float* qkv_b    = (const float*)d_in[5];
  const float* proj_w   = (const float*)d_in[6];
  const float* proj_b   = (const float*)d_in[7];
  const float* strength = (const float*)d_in[8];
  const float* n2g      = (const float*)d_in[9];
  const float* n2b      = (const float*)d_in[10];
  const float* w1       = (const float*)d_in[11];
  const float* b1       = (const float*)d_in[12];
  const float* w2       = (const float*)d_in[13];
  const float* b2       = (const float*)d_in[14];
  float* out = (float*)d_out;

  char* ws = (char*)d_ws;
  size_t off = 0;
  auto carve = [&](size_t bytes) -> char* {
    char* p = ws + off;
    off += (bytes + 255) & ~(size_t)255;
    return p;
  };
  __bf16* qkvw_bf  = (__bf16*)carve((size_t)C3 * DIMC * 2);
  __bf16* projw_bf = (__bf16*)carve((size_t)DIMC * DIMC * 2);
  __bf16* w1_bf    = (__bf16*)carve((size_t)HIDDEN * DIMC * 2);
  __bf16* w2_bf    = (__bf16*)carve((size_t)DIMC * HIDDEN * 2);
  __bf16* h_bf     = (__bf16*)carve((size_t)NTOK * DIMC * 2);
  __bf16* qkv_bf   = (__bf16*)carve((size_t)NTOK * C3 * 2);
  __bf16* attn_bf  = (__bf16*)carve((size_t)NTOK * DIMC * 2);
  float*  x1       = (float*)carve((size_t)NTOK * DIMC * 4);
  __bf16* h2_bf    = (__bf16*)carve((size_t)NTOK * DIMC * 2);
  __bf16* gelu_bf  = (__bf16*)carve((size_t)NTOK * HIDDEN * 2);

  // weight fp32 -> bf16
  cvt_f32_bf16<<<dim3(512), 256, 0, stream>>>(qkv_w, qkvw_bf, C3 * DIMC);
  cvt_f32_bf16<<<dim3(512), 256, 0, stream>>>(proj_w, projw_bf, DIMC * DIMC);
  cvt_f32_bf16<<<dim3(512), 256, 0, stream>>>(w1, w1_bf, HIDDEN * DIMC);
  cvt_f32_bf16<<<dim3(512), 256, 0, stream>>>(w2, w2_bf, DIMC * HIDDEN);

  // LN1 -> h
  layernorm_kernel<<<NTOK, 256, 0, stream>>>(x, n1g, n1b, h_bf);
  // qkv = h @ qkv_w^T + b
  gemm_wmma<0><<<dim3(NTOK / 128, C3 / 128), 256, 0, stream>>>(
      h_bf, qkvw_bf, qkv_b, nullptr, qkv_bf, NTOK, C3, DIMC);
  // masked flash attention
  flash_attn_kernel<<<dim3(NTOK / 128, HEADS), 256, 0, stream>>>(qkv_bf, elev, strength, attn_bf);
  // x1 = x + attn_out @ proj_w^T + b
  gemm_wmma<2><<<dim3(NTOK / 128, DIMC / 128), 256, 0, stream>>>(
      attn_bf, projw_bf, proj_b, x, x1, NTOK, DIMC, DIMC);
  // LN2 -> h2
  layernorm_kernel<<<NTOK, 256, 0, stream>>>(x1, n2g, n2b, h2_bf);
  // gelu(h2 @ w1^T + b1)
  gemm_wmma<1><<<dim3(NTOK / 128, HIDDEN / 128), 256, 0, stream>>>(
      h2_bf, w1_bf, b1, nullptr, gelu_bf, NTOK, HIDDEN, DIMC);
  // out = x1 + gelu @ w2^T + b2
  gemm_wmma<2><<<dim3(NTOK / 128, DIMC / 128), 256, 0, stream>>>(
      gelu_bf, w2_bf, b2, x1, out, NTOK, DIMC, HIDDEN);
}
